// ConditionalNSF_82179904242175
// MI455X (gfx1250) — matrix operand order repaired and checked by reading, not compiled, not run on hardware
//
#include <hip/hip_runtime.h>
#include <hip/hip_bf16.h>

typedef __attribute__((ext_vector_type(16))) _Float16 v16h;
typedef __attribute__((ext_vector_type(8)))  _Float16 v8h;
typedef __attribute__((ext_vector_type(8)))  float    v8f;

#define NTHR   256
#define MTILE  128
#define HIDN   128
#define DCND   48
#define KBINS  24
#define TSTEPS 10
#define NPARAM 73
#define PPAD   80
#define TAILV  15.0f

// f16 weight panel sizes (halves), [N][K] transposed + padded layouts
#define P0H (HIDN * 64)     // L0: Npad=128, Kpad=64
#define P1H (HIDN * HIDN)   // L1
#define P2H (HIDN * HIDN)   // L2
#define P3H (PPAD * HIDN)   // L3: Npad=80, Kpad=128
#define STEPH (P0H + P1H + P2H + P3H)   // 51200 halves / step

// ---- fragment loaders (CDNA5 WMMA 16x16x32 f16 VGPR layouts, ISA 7.12.2) ----

// A: 16x32 f16 tile, row-major, row stride `ld` halves. lane m = lane&15;
// halves 0..7 -> K=koff+0..7, halves 8..15 -> K=16+koff+0..7, koff=8*(lane>=16).
__device__ __forceinline__ v16h load_a_frag(const _Float16* rowbase, int ld, int kc, int lane) {
  const int m    = lane & 15;
  const int koff = (lane >> 4) << 3;
  const _Float16* p = rowbase + m * ld + kc * 32 + koff;
  v8h lo = *(const v8h*)(p);
  v8h hi = *(const v8h*)(p + 16);
  v16h a;
#pragma unroll
  for (int i = 0; i < 8; ++i) { a[i] = lo[i]; a[i + 8] = hi[i]; }
  return a;
}

// B: 32x16 tile from weights stored transposed [N][K] (stride kpad):
// lane n = lane&15; halves 0..7 -> K=kro+0..7, halves 8..15 -> K=16+kro+0..7, kro = 8*(lane>=16).
__device__ __forceinline__ v16h load_b_frag(const _Float16* wt, int kpad, int kc, int nt, int lane) {
  const int n   = lane & 15;
  const int kro = (lane >> 4) << 3;
  const _Float16* p = wt + (nt * 16 + n) * kpad + kc * 32 + kro;
  v8h lo = *(const v8h*)(p);
  v8h hi = *(const v8h*)(p + 16);
  v16h b;
#pragma unroll
  for (int i = 0; i < 8; ++i) { b[i] = lo[i]; b[i + 8] = hi[i]; }
  return b;
}

// Fallback staging: f32 global [Kdim][Ndim] -> LDS f16 transposed [Npad][Kpad]
__device__ __forceinline__ void stage_weights_f32(const float* __restrict__ Wg,
                                                  int Kdim, int Ndim, int Kpad, int Npad,
                                                  _Float16* wt) {
  for (int idx = threadIdx.x; idx < Npad * Kpad; idx += NTHR) {
    int nn = idx / Kpad, kk = idx - nn * Kpad;
    float v = (nn < Ndim && kk < Kdim) ? Wg[kk * Ndim + nn] : 0.0f;
    wt[idx] = (_Float16)v;
  }
}

// Async staging: pre-converted f16 panel in global -> LDS, 16B per op,
// tracked on ASYNCcnt (CDNA5 GLOBAL_LOAD_ASYNC_TO_LDS_B128).
__device__ __forceinline__ void issue_async_panel(const _Float16* __restrict__ src,
                                                  _Float16* dstLds, int halves) {
  const int nchunks = halves >> 3;                  // 16-byte chunks
  const unsigned long long gbase = (unsigned long long)(uintptr_t)src;
  const unsigned ldsbase = (unsigned)(uintptr_t)dstLds;   // low 32 bits = LDS byte addr
  for (int c = threadIdx.x; c < nchunks; c += NTHR) {
    unsigned off = (unsigned)c << 4;
    unsigned lds = ldsbase + off;
    asm volatile("global_load_async_to_lds_b128 %0, %1, %2"
                 :
                 : "v"(lds), "v"(off), "s"(gbase)
                 : "memory");
  }
}

__device__ __forceinline__ void wait_async_all() {
  asm volatile("s_wait_asynccnt 0x0" ::: "memory");
}

__device__ __forceinline__ void stage_bias(const float* __restrict__ bg, int Ndim, float* sb) {
  for (int nn = threadIdx.x; nn < HIDN; nn += NTHR)
    sb[nn] = (nn < Ndim) ? bg[nn] : 0.0f;
}

// One dense layer: wave wv computes rows [wv*16, wv*16+16). All A-frags loaded
// before any writeback, so in-place (in == outH) is safe per-wave.
template<int KCH, int NT, int LDIN, int KPAD, bool PARAMS>
__device__ __forceinline__ void layer_gemm(const _Float16* in, const _Float16* wt,
                                           const float* bias, _Float16* outH, float* outP) {
  const int lane = threadIdx.x & 31;
  const int wv   = threadIdx.x >> 5;
  const int m0   = wv * 16;

  v16h a[KCH];
#pragma unroll
  for (int kc = 0; kc < KCH; ++kc) a[kc] = load_a_frag(in + m0 * LDIN, LDIN, kc, lane);

  v8f acc[NT];
#pragma unroll
  for (int nt = 0; nt < NT; ++nt)
#pragma unroll
    for (int i = 0; i < 8; ++i) acc[nt][i] = 0.0f;

#pragma unroll
  for (int kc = 0; kc < KCH; ++kc) {
#pragma unroll
    for (int nt = 0; nt < NT; ++nt) {
      v16h b = load_b_frag(wt, KPAD, kc, nt, lane);
      acc[nt] = __builtin_amdgcn_wmma_f32_16x16x32_f16(
          false, a[kc], false, b, (short)0, acc[nt], false, false);
    }
  }

  // C/D layout: VGPR r, lane l -> M = r + 8*(l>=16), N = l&15 (per 16-tile).
  const int nl = lane & 15;
  const int mo = (lane >> 4) * 8;
#pragma unroll
  for (int nt = 0; nt < NT; ++nt) {
    const int n = nt * 16 + nl;
    const float bv = bias[n];
#pragma unroll
    for (int r = 0; r < 8; ++r) {
      const int m = m0 + mo + r;
      float v = acc[nt][r] + bv;
      if (PARAMS) {
        outP[m * PPAD + n] = v;
      } else {
        // SiLU with single v_rcp_f32 (native reciprocal), not IEEE div expansion
        float sg = __builtin_amdgcn_rcpf(1.0f + __expf(-v));
        outH[m * HIDN + n] = (_Float16)(v * sg);
      }
    }
  }
}

template<bool ASYNCW>
__global__ __launch_bounds__(NTHR, 1)
void nsf_fused_kernel(const float* __restrict__ z,  const float* __restrict__ ctx,
                      const float* __restrict__ W0, const float* __restrict__ b0,
                      const float* __restrict__ W1, const float* __restrict__ b1,
                      const float* __restrict__ W2, const float* __restrict__ b2,
                      const float* __restrict__ W3, const float* __restrict__ b3,
                      const float* __restrict__ loc, const float* __restrict__ scale,
                      const _Float16* __restrict__ wpan,
                      float* __restrict__ out, int n) {
  __shared__ __align__(16) _Float16 sC[MTILE * 64];
  __shared__ __align__(16) _Float16 sH[MTILE * HIDN];
  __shared__ __align__(16) _Float16 sW[2][HIDN * HIDN];   // double-buffered weights
  __shared__ float sB[2][HIDN];
  __shared__ float sP[MTILE * PPAD];
  __shared__ float sZ[MTILE];
  __shared__ float sLD[MTILE];

  const int row0 = blockIdx.x * MTILE;

  // --- stage context (f16, K padded 48->64) + carries ---
  for (int idx = threadIdx.x; idx < MTILE * 64; idx += NTHR) {
    int r = idx >> 6, k = idx & 63;
    int row = row0 + r;
    float v = (row < n && k < DCND) ? ctx[row * DCND + k] : 0.0f;
    sC[idx] = (_Float16)v;
  }
  if (threadIdx.x < MTILE) {
    int row = row0 + threadIdx.x;
    sZ[threadIdx.x]  = (row < n) ? z[row] : 0.0f;
    sLD[threadIdx.x] = 0.0f;
  }

  // --- prologue: stage first weight panel (t=0, layer 0) into buffer 0 ---
  if (ASYNCW) issue_async_panel(wpan, sW[0], P0H);
  else        stage_weights_f32(W0, DCND, HIDN, 64, HIDN, sW[0]);
  stage_bias(b0, HIDN, sB[0]);
  if (ASYNCW) wait_async_all();
  __syncthreads();

  int cur = 0;
  for (int t = 0; t < TSTEPS; ++t) {
    const _Float16* step = wpan + (size_t)t * STEPH;

    // ---- L0: stage L1 panel into other buffer, compute L0 ----
    if (ASYNCW) issue_async_panel(step + P0H, sW[cur ^ 1], P1H);
    else        stage_weights_f32(W1 + t * HIDN * HIDN, HIDN, HIDN, HIDN, HIDN, sW[cur ^ 1]);
    stage_bias(b1 + t * HIDN, HIDN, sB[cur ^ 1]);
    layer_gemm<2, 8, 64, 64, false>(sC, sW[cur], sB[cur], sH, nullptr);
    if (ASYNCW) wait_async_all();
    __syncthreads();
    cur ^= 1;

    // ---- L1: stage L2, compute L1 (in-place) ----
    if (ASYNCW) issue_async_panel(step + P0H + P1H, sW[cur ^ 1], P2H);
    else        stage_weights_f32(W2 + t * HIDN * HIDN, HIDN, HIDN, HIDN, HIDN, sW[cur ^ 1]);
    stage_bias(b2 + t * HIDN, HIDN, sB[cur ^ 1]);
    layer_gemm<4, 8, HIDN, HIDN, false>(sH, sW[cur], sB[cur], sH, nullptr);
    if (ASYNCW) wait_async_all();
    __syncthreads();
    cur ^= 1;

    // ---- L2: stage L3, compute L2 ----
    if (ASYNCW) issue_async_panel(step + P0H + P1H + P2H, sW[cur ^ 1], P3H);
    else        stage_weights_f32(W3 + t * HIDN * NPARAM, HIDN, NPARAM, HIDN, PPAD, sW[cur ^ 1]);
    stage_bias(b3 + t * NPARAM, NPARAM, sB[cur ^ 1]);
    layer_gemm<4, 8, HIDN, HIDN, false>(sH, sW[cur], sB[cur], sH, nullptr);
    if (ASYNCW) wait_async_all();
    __syncthreads();
    cur ^= 1;

    // ---- L3: stage next step's L0 (if any), compute params ----
    if (t + 1 < TSTEPS) {
      if (ASYNCW) issue_async_panel(wpan + (size_t)(t + 1) * STEPH, sW[cur ^ 1], P0H);
      else        stage_weights_f32(W0 + (t + 1) * DCND * HIDN, DCND, HIDN, 64, HIDN, sW[cur ^ 1]);
      stage_bias(b0 + (t + 1) * HIDN, HIDN, sB[cur ^ 1]);
    }
    layer_gemm<4, 5, HIDN, HIDN, true>(sH, sW[cur], sB[cur], nullptr, sP);
    if (ASYNCW) wait_async_all();
    __syncthreads();
    cur ^= 1;

    // ---- rational-quadratic spline epilogue: one thread per row ----
    if (threadIdx.x < MTILE) {
      const float* p = sP + threadIdx.x * PPAD;
      float x = sZ[threadIdx.x];

      float mw = -3.0e38f, mh = -3.0e38f;
#pragma unroll
      for (int j = 0; j < KBINS; ++j) {
        mw = fmaxf(mw, p[j]);
        mh = fmaxf(mh, p[KBINS + j]);
      }
      float ew[KBINS], eh[KBINS];
      float swm = 0.f, shm = 0.f;
#pragma unroll
      for (int j = 0; j < KBINS; ++j) {
        ew[j] = __expf(p[j] - mw);         swm += ew[j];
        eh[j] = __expf(p[KBINS + j] - mh); shm += eh[j];
      }
      const float iw = (2.0f * TAILV) / swm;
      const float ih = (2.0f * TAILV) / shm;

      int bidx = 0;
      float cum = -TAILV;
#pragma unroll
      for (int j = 0; j < KBINS; ++j) {
        cum += ew[j] * iw;
        if (cum < x && bidx < KBINS - 1) bidx++;
      }
      float wk = 0.f, cwk = -TAILV, hk = 0.f, chk = -TAILV;
#pragma unroll
      for (int j = 0; j < KBINS; ++j) {
        float wj = ew[j] * iw, hj = eh[j] * ih;
        cwk += (j < bidx) ? wj : 0.f;
        chk += (j < bidx) ? hj : 0.f;
        wk = (j == bidx) ? wj : wk;
        hk = (j == bidx) ? hj : hk;
      }
      float d0 = p[2 * KBINS + bidx];
      float d1 = p[2 * KBINS + bidx + 1];
      float dk  = fmaxf(d0, 0.f) + __logf(1.f + __expf(-fabsf(d0))) + 0.001f;  // softplus
      float dk1 = fmaxf(d1, 0.f) + __logf(1.f + __expf(-fabsf(d1))) + 0.001f;

      float s  = hk / wk;
      float xi = (x - cwk) / wk;
      xi = fminf(fmaxf(xi, 1e-6f), 1.0f - 1e-6f);
      float om = 1.0f - xi;
      float numer = hk * (s * xi * xi + dk * xi * om);
      float den   = s + (dk + dk1 - 2.0f * s) * xi * om;
      float y  = chk + numer / den;
      float dy = s * s * (dk1 * xi * xi + 2.0f * s * xi * om + dk * om * om) / (den * den);
      float ldet = __logf(fmaxf(dy, 1e-10f)) + __logf(fmaxf(hk, 1e-10f)) - __logf(fmaxf(wk, 1e-10f));

      bool inside = (x >= -TAILV) && (x <= TAILV);
      sZ[threadIdx.x]   = inside ? y : x;
      sLD[threadIdx.x] += inside ? ldet : 0.0f;
    }
    __syncthreads();
  }

  if (threadIdx.x < MTILE) {
    int row = row0 + threadIdx.x;
    if (row < n) {
      float sc = fmaxf(scale[0], 0.1f);
      out[row]     = sZ[threadIdx.x] * sc + loc[0];
      out[n + row] = sLD[threadIdx.x];
    }
  }
}

// Pre-convert all weights once: f32 [K][N] -> f16 transposed [Npad][Kpad] panels
// packed per-step in d_ws. Weights then live in L2 (192 MB) as f16.
__global__ void prep_weights_kernel(const float* __restrict__ W0, const float* __restrict__ W1,
                                    const float* __restrict__ W2, const float* __restrict__ W3,
                                    _Float16* __restrict__ ws) {
  const int total = TSTEPS * STEPH;
  for (int idx = blockIdx.x * blockDim.x + threadIdx.x; idx < total;
       idx += gridDim.x * blockDim.x) {
    int t = idx / STEPH;
    int r = idx - t * STEPH;
    const float* Wg; int Kdim, Ndim, Kpad;
    if (r < P0H) {
      Wg = W0 + t * DCND * HIDN;   Kdim = DCND; Ndim = HIDN;   Kpad = 64;
    } else if (r < P0H + P1H) {
      r -= P0H;
      Wg = W1 + t * HIDN * HIDN;   Kdim = HIDN; Ndim = HIDN;   Kpad = HIDN;
    } else if (r < P0H + P1H + P2H) {
      r -= P0H + P1H;
      Wg = W2 + t * HIDN * HIDN;   Kdim = HIDN; Ndim = HIDN;   Kpad = HIDN;
    } else {
      r -= P0H + P1H + P2H;
      Wg = W3 + t * HIDN * NPARAM; Kdim = HIDN; Ndim = NPARAM; Kpad = HIDN;
    }
    int nn = r / Kpad, kk = r - nn * Kpad;
    float v = (nn < Ndim && kk < Kdim) ? Wg[kk * Ndim + nn] : 0.0f;
    ws[idx] = (_Float16)v;
  }
}

extern "C" void kernel_launch(void* const* d_in, const int* in_sizes, int n_in,
                              void* d_out, int out_size, void* d_ws, size_t ws_size,
                              hipStream_t stream) {
  (void)n_in; (void)out_size;
  const float* z     = (const float*)d_in[0];
  const float* ctx   = (const float*)d_in[1];
  const float* W0    = (const float*)d_in[2];
  const float* b0    = (const float*)d_in[3];
  const float* W1    = (const float*)d_in[4];
  const float* b1    = (const float*)d_in[5];
  const float* W2    = (const float*)d_in[6];
  const float* b2    = (const float*)d_in[7];
  const float* W3    = (const float*)d_in[8];
  const float* b3    = (const float*)d_in[9];
  const float* loc   = (const float*)d_in[10];
  const float* scale = (const float*)d_in[11];
  float* out = (float*)d_out;
  const int n = in_sizes[0];
  const int nblocks = (n + MTILE - 1) / MTILE;

  const size_t need = (size_t)TSTEPS * STEPH * sizeof(_Float16);
  if (d_ws != nullptr && ws_size >= need) {
    _Float16* wpan = (_Float16*)d_ws;
    prep_weights_kernel<<<256, NTHR, 0, stream>>>(W0, W1, W2, W3, wpan);
    nsf_fused_kernel<true><<<nblocks, NTHR, 0, stream>>>(
        z, ctx, W0, b0, W1, b1, W2, b2, W3, b3, loc, scale, wpan, out, n);
  } else {
    nsf_fused_kernel<false><<<nblocks, NTHR, 0, stream>>>(
        z, ctx, W0, b0, W1, b1, W2, b2, W3, b3, loc, scale, nullptr, out, n);
  }
}
